// Eva_50148038148108
// MI455X (gfx1250) — matrix operand order repaired and checked by reading, not compiled, not run on hardware
//
#include <hip/hip_runtime.h>
#include <hip/hip_bf16.h>

typedef __attribute__((ext_vector_type(16))) _Float16 v16h;
typedef __attribute__((ext_vector_type(8)))  _Float16 v8h;
typedef __attribute__((ext_vector_type(8)))  float    v8f;

#define BKK 32

// ---------------------------------------------------------------------------
// CDNA5 async global->LDS copy (16B per lane), tracked by ASYNCcnt.
// VDST = per-lane LDS byte address, VADDR = 64-bit global address.
// LDS byte address = low 32 bits of the generic shared-aperture pointer.
// ---------------------------------------------------------------------------
__device__ __forceinline__ void async_copy_b128(unsigned lds_addr, const void* gaddr) {
    asm volatile("global_load_async_to_lds_b128 %0, %1, off"
                 :: "v"(lds_addr), "v"((unsigned long long)(size_t)gaddr)
                 : "memory");
}
__device__ __forceinline__ void wait_async0() {
    asm volatile("s_wait_asynccnt 0" ::: "memory");
}

// ---------------------------------------------------------------------------
// Weight convert + transpose: src f32 [L][O][I] -> dst f16 [L][I][O]
// ---------------------------------------------------------------------------
__global__ __launch_bounds__(256)
void transpose_w_kernel(const float* __restrict__ src, _Float16* __restrict__ dst,
                        int O, int I) {
    const long long layer = blockIdx.y;
    const long long base = layer * (long long)O * I;
    long long idx = (long long)blockIdx.x * 256 + threadIdx.x;
    if (idx >= (long long)O * I) return;
    int o = (int)(idx / I);
    int i = (int)(idx % I);
    dst[base + (long long)i * O + o] = (_Float16)src[base + idx];
}

// ---------------------------------------------------------------------------
// x + pos_embed (broadcast over batch) -> out (residual stream h)
// ---------------------------------------------------------------------------
__global__ __launch_bounds__(256)
void add_pos_kernel(const float* __restrict__ x, const float* __restrict__ pos,
                    float* __restrict__ out) {
    long long idx = (long long)blockIdx.x * 256 + threadIdx.x;
    out[idx] = x[idx] + pos[idx % 196608];   // 256*768
}

// ---------------------------------------------------------------------------
// Row LayerNorm f32 -> f16.  One 256-thread block per row, C in {768, 2048}.
// ---------------------------------------------------------------------------
__global__ __launch_bounds__(256)
void ln_kernel(const float* __restrict__ x, const float* __restrict__ w,
               const float* __restrict__ b, _Float16* __restrict__ out, int C) {
    const long long row = blockIdx.x;
    const float* xr = x + row * C;
    _Float16* orow = out + row * C;
    float vals[8];
    const int n = C >> 8;
    float s = 0.f, s2 = 0.f;
    for (int j = 0; j < n; ++j) {
        float v = xr[threadIdx.x + (j << 8)];
        vals[j] = v; s += v; s2 += v * v;
    }
    __shared__ float rs[256];
    __shared__ float rq[256];
    rs[threadIdx.x] = s; rq[threadIdx.x] = s2;
    __syncthreads();
    for (int off = 128; off > 0; off >>= 1) {
        if (threadIdx.x < (unsigned)off) {
            rs[threadIdx.x] += rs[threadIdx.x + off];
            rq[threadIdx.x] += rq[threadIdx.x + off];
        }
        __syncthreads();
    }
    float mean = rs[0] / C;
    float var  = rq[0] / C - mean * mean;
    float rstd = rsqrtf(var + 1e-5f);
    for (int j = 0; j < n; ++j) {
        int col = threadIdx.x + (j << 8);
        orow[col] = (_Float16)((vals[j] - mean) * rstd * w[col] + b[col]);
    }
}

// ---------------------------------------------------------------------------
// SiLU(g)*u followed by LayerNorm over 2048 -> f16
// ---------------------------------------------------------------------------
__global__ __launch_bounds__(256)
void swiglu_ln_kernel(const float* __restrict__ g, const float* __restrict__ u,
                      const float* __restrict__ w, const float* __restrict__ b,
                      _Float16* __restrict__ out) {
    const long long row = blockIdx.x;
    const float* gr = g + row * 2048;
    const float* ur = u + row * 2048;
    _Float16* orow = out + row * 2048;
    float vals[8];
    float s = 0.f, s2 = 0.f;
    for (int j = 0; j < 8; ++j) {
        int col = threadIdx.x + (j << 8);
        float gv = gr[col];
        float t = (gv / (1.f + __expf(-gv))) * ur[col];
        vals[j] = t; s += t; s2 += t * t;
    }
    __shared__ float rs[256];
    __shared__ float rq[256];
    rs[threadIdx.x] = s; rq[threadIdx.x] = s2;
    __syncthreads();
    for (int off = 128; off > 0; off >>= 1) {
        if (threadIdx.x < (unsigned)off) {
            rs[threadIdx.x] += rs[threadIdx.x + off];
            rq[threadIdx.x] += rq[threadIdx.x + off];
        }
        __syncthreads();
    }
    float mean = rs[0] * (1.f / 2048.f);
    float var  = rq[0] * (1.f / 2048.f) - mean * mean;
    float rstd = rsqrtf(var + 1e-5f);
    for (int j = 0; j < 8; ++j) {
        int col = threadIdx.x + (j << 8);
        orow[col] = (_Float16)((vals[j] - mean) * rstd * w[col] + b[col]);
    }
}

// ---------------------------------------------------------------------------
// RoPE on q,k, scale q by 1/sqrt(64); emit per-(b,h) f16 tiles:
//   qh [bh][n][d] (scaled), kth [bh][d][n] (transposed), vh [bh][n][d]
// ---------------------------------------------------------------------------
__global__ __launch_bounds__(256)
void rope_prep_kernel(const float* __restrict__ q, const float* __restrict__ k,
                      const float* __restrict__ v, const float* __restrict__ rope,
                      _Float16* __restrict__ qh, _Float16* __restrict__ kth,
                      _Float16* __restrict__ vh) {
    long long idx = (long long)blockIdx.x * 256 + threadIdx.x;  // over 4096*768
    int col = (int)(idx % 768);
    long long row = idx / 768;           // b*256 + n
    int n = (int)(row & 255);
    int bb = (int)(row >> 8);
    int h = col >> 6;
    int d = col & 63;
    float sn = rope[n * 128 + d];
    float cs = rope[n * 128 + 64 + d];
    float qv = q[idx], kv = k[idx];
    float qp = (d & 1) ? q[idx - 1] : -q[idx + 1];
    float kp = (d & 1) ? k[idx - 1] : -k[idx + 1];
    long long bh = (long long)bb * 12 + h;
    qh [bh * 16384 + n * 64 + d]  = (_Float16)((qv * cs + qp * sn) * 0.125f);
    kth[bh * 16384 + d * 256 + n] = (_Float16)(kv * cs + kp * sn);
    vh [bh * 16384 + n * 64 + d]  = (_Float16)v[idx];
}

// ---------------------------------------------------------------------------
// Softmax, one wave32 per 256-wide row. f32 in, f16 probs out.
// ---------------------------------------------------------------------------
__global__ __launch_bounds__(256)
void softmax_kernel(const float* __restrict__ S, _Float16* __restrict__ P) {
    int lane = threadIdx.x & 31;
    long long row = (long long)blockIdx.x * 8 + (threadIdx.x >> 5);
    const float* sr = S + row * 256;
    float v[8], mx = -3.4e38f;
    for (int j = 0; j < 8; ++j) { v[j] = sr[lane + j * 32]; mx = fmaxf(mx, v[j]); }
    for (int o = 16; o > 0; o >>= 1) mx = fmaxf(mx, __shfl_xor(mx, o, 32));
    float s = 0.f;
    for (int j = 0; j < 8; ++j) { v[j] = __expf(v[j] - mx); s += v[j]; }
    for (int o = 16; o > 0; o >>= 1) s += __shfl_xor(s, o, 32);
    float inv = 1.f / s;
    _Float16* pr = P + row * 256;
    for (int j = 0; j < 8; ++j) pr[lane + j * 32] = (_Float16)(v[j] * inv);
}

// ---------------------------------------------------------------------------
// [bh][n][d] f32 -> [b*256+n][h*64+d] f16
// ---------------------------------------------------------------------------
__global__ __launch_bounds__(256)
void reshape_o_kernel(const float* __restrict__ Obh, _Float16* __restrict__ oh) {
    long long idx = (long long)blockIdx.x * 256 + threadIdx.x;  // 192*256*64
    int d = (int)(idx & 63);
    long long t = idx >> 6;
    int n = (int)(t & 255);
    long long bh = t >> 8;
    int h = (int)(bh % 12);
    int bb = (int)(bh / 12);
    oh[((long long)(bb * 256 + n)) * 768 + h * 64 + d] = (_Float16)Obh[idx];
}

// ---------------------------------------------------------------------------
// WMMA f16 GEMM: C[M][N] = A[M][K](f16) x Bt[K][N](f16) (+bias) (+resid), f32 out.
// WROWSxWCOLS wave32 grid; per-wave (TBM/WROWS)x(TBN/WCOLS) output as MFxNF
// 16x16 fragments.  Double-buffered LDS filled with
// global_load_async_to_lds_b128 (ASYNCcnt), one barrier per K-step.
// Big variant: 128 thr, 128x128 tile, 2x2 waves, 4x4 frags -> 1.0 ds-load/WMMA.
// ---------------------------------------------------------------------------
template<int THREADS, int TBM, int TBN, int WROWS, int WCOLS,
         bool HAS_BIAS, bool ADD_RESID>
__global__ __launch_bounds__(THREADS)
void gemm_f16_kernel(const _Float16* __restrict__ A, long long aBatch, int lda,
                     const _Float16* __restrict__ B, long long bBatch, int ldb,
                     const float* __restrict__ bias,
                     const float* __restrict__ resid, long long rBatch,
                     float* __restrict__ C, long long cBatch, int ldc, int K) {
    constexpr int MF = (TBM / WROWS) / 16;
    constexpr int NF = (TBN / WCOLS) / 16;
    constexpr int ASEG = (TBM * 2) / THREADS;   // 16-half segments per thread (A)
    constexpr int BSEG = (TBN * 2) / THREADS;   // 16-half segments per thread (B)
    constexpr int BROWSEG = TBN / 16;           // segments per B row
    __shared__ _Float16 As[2][TBM][BKK + 8];
    __shared__ _Float16 Bs[2][BKK][TBN + 8];

    const int tid  = threadIdx.x;
    const int lane = tid & 31;
    const int wave = tid >> 5;
    const int m0 = blockIdx.y * TBM;
    const int n0 = blockIdx.x * TBN;
    const long long batch = blockIdx.z;
    A += batch * aBatch;
    B += batch * bBatch;
    C += batch * cBatch;
    if (ADD_RESID) resid += batch * rBatch;

    const int wm = (wave / WCOLS) * (TBM / WROWS);
    const int wn = (wave % WCOLS) * (TBN / WCOLS);

    v8f acc[MF][NF];
#pragma unroll
    for (int i = 0; i < MF; ++i)
#pragma unroll
        for (int j = 0; j < NF; ++j)
#pragma unroll
            for (int e = 0; e < 8; ++e) acc[i][j][e] = 0.f;

    auto issue = [&](int buf, int k0) {
#pragma unroll
        for (int s = 0; s < ASEG; ++s) {
            const int seg = tid + s * THREADS;         // A row = 2 segments
            const int r = seg >> 1, c = (seg & 1) * 16;
            const _Float16* ag = A + (long long)(m0 + r) * lda + (k0 + c);
            unsigned la = (unsigned)(size_t)&As[buf][r][c];
            async_copy_b128(la,      ag);
            async_copy_b128(la + 16, ag + 8);
        }
#pragma unroll
        for (int s = 0; s < BSEG; ++s) {
            const int seg = tid + s * THREADS;         // B row = TBN/16 segments
            const int r = seg / BROWSEG, c = (seg % BROWSEG) * 16;
            const _Float16* bg = B + (long long)(k0 + r) * ldb + (n0 + c);
            unsigned lb = (unsigned)(size_t)&Bs[buf][r][c];
            async_copy_b128(lb,      bg);
            async_copy_b128(lb + 16, bg + 8);
        }
    };

    issue(0, 0);
    wait_async0();
    __syncthreads();

    const int am = lane & 15;
    const int kb = (lane & 16) ? 8 : 0;   // ISA 16-bit A layout: K split by lane half
    const int nk = K / BKK;

    for (int kk = 0; kk < nk; ++kk) {
        const int cur = kk & 1;
        if (kk + 1 < nk) issue(cur ^ 1, (kk + 1) * BKK);

        v16h afrag[MF];
#pragma unroll
        for (int i = 0; i < MF; ++i) {
            const _Float16* ap = &As[cur][wm + i * 16 + am][kb];
            v8h alo = *(const v8h*)ap;
            v8h ahi = *(const v8h*)(ap + 16);
            afrag[i] = __builtin_shufflevector(alo, ahi, 0, 1, 2, 3, 4, 5, 6, 7,
                                               8, 9, 10, 11, 12, 13, 14, 15);
        }
#pragma unroll
        for (int j = 0; j < NF; ++j) {
            const _Float16* bp = &Bs[cur][kb][wn + j * 16 + am];   // N = lane&15, K = kb+s / kb+16+s
            v16h bfrag;
#pragma unroll
            for (int s = 0; s < 8; ++s) {
                bfrag[s]     = bp[s * (TBN + 8)];
                bfrag[s + 8] = bp[(s + 16) * (TBN + 8)];
            }
#pragma unroll
            for (int i = 0; i < MF; ++i)
                acc[i][j] = __builtin_amdgcn_wmma_f32_16x16x32_f16(
                    false, afrag[i], false, bfrag, (short)0, acc[i][j],
                    false, false);
        }
        wait_async0();
        __syncthreads();
    }

    // Epilogue per ISA C/D layout: VGPR v -> M = v + 8*(lane>=16), N = lane&15
    const int rsel = (lane >> 4) << 3;
    const int cn = lane & 15;
#pragma unroll
    for (int i = 0; i < MF; ++i)
#pragma unroll
        for (int j = 0; j < NF; ++j) {
            const int colBase = n0 + wn + j * 16 + cn;
            const float bb = HAS_BIAS ? bias[colBase] : 0.f;
#pragma unroll
            for (int v = 0; v < 8; ++v) {
                const int row = m0 + wm + i * 16 + v + rsel;
                float val = acc[i][j][v] + bb;
                if (ADD_RESID) val += resid[(long long)row * ldc + colBase];
                C[(long long)row * ldc + colBase] = val;
            }
        }
}

// ---------------------------------------------------------------------------
// Host orchestration
// ---------------------------------------------------------------------------
extern "C" void kernel_launch(void* const* d_in, const int* in_sizes, int n_in,
                              void* d_out, int out_size, void* d_ws, size_t ws_size,
                              hipStream_t stream) {
    const float* x        = (const float*)d_in[0];
    const float* pos      = (const float*)d_in[1];
    const float* rope     = (const float*)d_in[2];
    const float* ln1_w    = (const float*)d_in[3];
    const float* ln1_b    = (const float*)d_in[4];
    const float* wq       = (const float*)d_in[5];
    const float* bq       = (const float*)d_in[6];
    const float* wk       = (const float*)d_in[7];
    const float* wv       = (const float*)d_in[8];
    const float* bv       = (const float*)d_in[9];
    const float* wo       = (const float*)d_in[10];
    const float* bo       = (const float*)d_in[11];
    const float* ln2_w    = (const float*)d_in[12];
    const float* ln2_b    = (const float*)d_in[13];
    const float* w1g      = (const float*)d_in[14];
    const float* b1g      = (const float*)d_in[15];
    const float* w1x      = (const float*)d_in[16];
    const float* b1x      = (const float*)d_in[17];
    const float* lnm_w    = (const float*)d_in[18];
    const float* lnm_b    = (const float*)d_in[19];
    const float* w2       = (const float*)d_in[20];
    const float* b2       = (const float*)d_in[21];
    float* out = (float*)d_out;   // residual stream h lives here

    char* ws = (char*)d_ws;
    size_t off = 0;
    auto carve = [&](size_t bytes) -> void* {
        void* p = ws + off;
        off = (off + bytes + 255) & ~(size_t)255;
        return p;
    };

    const size_t WQK = 12ULL * 768 * 768;
    const size_t WH  = 12ULL * 768 * 2048;
    _Float16* wtq  = (_Float16*)carve(WQK * 2);
    _Float16* wtk  = (_Float16*)carve(WQK * 2);
    _Float16* wtv  = (_Float16*)carve(WQK * 2);
    _Float16* wto  = (_Float16*)carve(WQK * 2);
    _Float16* wt1g = (_Float16*)carve(WH * 2);
    _Float16* wt1x = (_Float16*)carve(WH * 2);
    _Float16* wt2  = (_Float16*)carve(WH * 2);
    _Float16* y16  = (_Float16*)carve(4096ULL * 2048 * 2);
    float*    qf   = (float*)carve(4096ULL * 768 * 4);
    float*    kf   = (float*)carve(4096ULL * 768 * 4);
    float*    vf   = (float*)carve(4096ULL * 768 * 4);
    float*    gf   = (float*)carve(4096ULL * 2048 * 4);
    float*    uf   = (float*)carve(4096ULL * 2048 * 4);
    _Float16* qh   = (_Float16*)carve(192ULL * 16384 * 2);
    _Float16* kth  = (_Float16*)carve(192ULL * 16384 * 2);
    _Float16* vh   = (_Float16*)carve(192ULL * 16384 * 2);
    float*    Sf   = (float*)carve(192ULL * 65536 * 4);
    _Float16* Ph   = (_Float16*)carve(192ULL * 65536 * 2);
    float*    Obh  = (float*)carve(192ULL * 16384 * 4);
    _Float16* oh   = (_Float16*)carve(4096ULL * 768 * 2);

    transpose_w_kernel<<<dim3(2304, 12), 256, 0, stream>>>(wq,  wtq,  768,  768);
    transpose_w_kernel<<<dim3(2304, 12), 256, 0, stream>>>(wk,  wtk,  768,  768);
    transpose_w_kernel<<<dim3(2304, 12), 256, 0, stream>>>(wv,  wtv,  768,  768);
    transpose_w_kernel<<<dim3(2304, 12), 256, 0, stream>>>(wo,  wto,  768,  768);
    transpose_w_kernel<<<dim3(6144, 12), 256, 0, stream>>>(w1g, wt1g, 2048, 768);
    transpose_w_kernel<<<dim3(6144, 12), 256, 0, stream>>>(w1x, wt1x, 2048, 768);
    transpose_w_kernel<<<dim3(6144, 12), 256, 0, stream>>>(w2,  wt2,  768, 2048);

    add_pos_kernel<<<12288, 256, 0, stream>>>(x, pos, out);

    for (int l = 0; l < 12; ++l) {
        const long long oQ = (long long)l * 768 * 768;
        const long long oH = (long long)l * 768 * 2048;

        // ---- attention ----
        ln_kernel<<<4096, 256, 0, stream>>>(out, ln1_w + l * 768, ln1_b + l * 768,
                                            y16, 768);
        gemm_f16_kernel<128, 128, 128, 2, 2, true,  false>
            <<<dim3(6, 32, 1), 128, 0, stream>>>(
            y16, 0, 768, wtq + oQ, 0, 768, bq + l * 768, nullptr, 0,
            qf, 0, 768, 768);
        gemm_f16_kernel<128, 128, 128, 2, 2, false, false>
            <<<dim3(6, 32, 1), 128, 0, stream>>>(
            y16, 0, 768, wtk + oQ, 0, 768, nullptr, nullptr, 0,
            kf, 0, 768, 768);
        gemm_f16_kernel<128, 128, 128, 2, 2, true,  false>
            <<<dim3(6, 32, 1), 128, 0, stream>>>(
            y16, 0, 768, wtv + oQ, 0, 768, bv + l * 768, nullptr, 0,
            vf, 0, 768, 768);
        rope_prep_kernel<<<12288, 256, 0, stream>>>(qf, kf, vf, rope, qh, kth, vh);
        // scores: [bh] 256x256 = Q(256x64) x Kt(64x256), scale folded into qh
        gemm_f16_kernel<128, 128, 128, 2, 2, false, false>
            <<<dim3(2, 2, 192), 128, 0, stream>>>(
            qh, 16384, 64, kth, 16384, 256, nullptr, nullptr, 0,
            Sf, 65536, 256, 64);
        softmax_kernel<<<6144, 256, 0, stream>>>(Sf, Ph);
        // O: [bh] 256x64 = P(256x256) x V(256x64)  (64-wide N -> small tile)
        gemm_f16_kernel<128, 64, 64, 2, 2, false, false>
            <<<dim3(1, 4, 192), 128, 0, stream>>>(
            Ph, 65536, 256, vh, 16384, 64, nullptr, nullptr, 0,
            Obh, 16384, 64, 256);
        reshape_o_kernel<<<12288, 256, 0, stream>>>(Obh, oh);
        // h = h + O @ Wo^T + bo
        gemm_f16_kernel<128, 128, 128, 2, 2, true, true>
            <<<dim3(6, 32, 1), 128, 0, stream>>>(
            oh, 0, 768, wto + oQ, 0, 768, bo + l * 768, out, 0,
            out, 0, 768, 768);

        // ---- SwiGLU MLP with inner LN ----
        ln_kernel<<<4096, 256, 0, stream>>>(out, ln2_w + l * 768, ln2_b + l * 768,
                                            y16, 768);
        gemm_f16_kernel<128, 128, 128, 2, 2, true, false>
            <<<dim3(16, 32, 1), 128, 0, stream>>>(
            y16, 0, 768, wt1g + oH, 0, 2048, b1g + l * 2048, nullptr, 0,
            gf, 0, 2048, 768);
        gemm_f16_kernel<128, 128, 128, 2, 2, true, false>
            <<<dim3(16, 32, 1), 128, 0, stream>>>(
            y16, 0, 768, wt1x + oH, 0, 2048, b1x + l * 2048, nullptr, 0,
            uf, 0, 2048, 768);
        swiglu_ln_kernel<<<4096, 256, 0, stream>>>(gf, uf, lnm_w + l * 2048,
                                                   lnm_b + l * 2048, y16);
        // h = h + m @ W2^T + b2
        gemm_f16_kernel<128, 128, 128, 2, 2, true, true>
            <<<dim3(6, 32, 1), 128, 0, stream>>>(
            y16, 0, 2048, wt2 + oH, 0, 768, b2 + l * 768, out, 0,
            out, 0, 768, 768);
    }

    (void)in_sizes; (void)n_in; (void)out_size; (void)ws_size;
}
